// Net_45165876085093
// MI455X (gfx1250) — matrix operand order repaired and checked by reading, not compiled, not run on hardware
//
#include <hip/hip_runtime.h>

// ---------------- problem constants (match reference) ----------------
#define PN   50000
#define HD   128
#define ED   600000
#define BB   256
#define LL   50
#define LIN  30
#define NCLS 5
#define H3   (3*HD)   // 384
#define H4   (4*HD)   // 512
#define BL   (BB*LL)  // 12800

typedef __attribute__((ext_vector_type(16))) _Float16 v16h;
typedef __attribute__((ext_vector_type(8)))  _Float16 v8h;
typedef __attribute__((ext_vector_type(4)))  _Float16 v4h;
typedef __attribute__((ext_vector_type(8)))  float    v8f;

// ---------------- workspace layout ----------------
// f32 region (offsets in floats)
static const size_t OFF_H    = 0;                                // P*H
static const size_t OFF_MSG  = OFF_H   + (size_t)PN*HD;          // P*H
static const size_t OFF_AGG  = OFF_MSG + (size_t)PN*HD;          // P*H
static const size_t OFF_GI   = OFF_AGG + (size_t)PN*HD;          // P*3H (reused: XW, G)
static const size_t OFF_GH   = OFF_GI  + (size_t)PN*H3;          // P*3H (reused: LSTM/fc tensors)
static const size_t OFF_W    = OFF_GH  + (size_t)PN*H3;
// aliases inside OFF_GI region
static const size_t OFF_XW   = OFF_GI;                           // BL*H4
static const size_t OFF_G    = OFF_GI + 7000000;                 // B*H4
// aliases inside OFF_GH region
static const size_t OFF_UH   = OFF_GH;                           // BL*H
static const size_t OFF_REG  = OFF_GH + 2000000;
static const size_t OFF_POI  = OFF_GH + 2100000;
static const size_t OFF_HS   = OFF_GH + 2200000;
static const size_t OFF_CS   = OFF_GH + 2300000;
static const size_t OFF_URP  = OFF_GH + 2400000;
static const size_t OFF_URP2 = OFF_GH + 2500000;
static const size_t OFF_UPP  = OFF_GH + 2600000;
static const size_t OFF_UPP2 = OFF_GH + 2700000;
static const size_t OFF_LOG  = OFF_GH + 2800000;
// small f32 scratch
static const size_t OFF_SCALE = OFF_W;                           // 512
static const size_t OFF_SHIFT = OFF_W + 512;                     // 512
static const size_t OFF_F16   = OFF_W + 1024;                    // halves region starts here (16B aligned)
// f16 region (offsets in halves, relative to f16 base); +2048 = 16 rows tail pad
static const size_t HF_H    = 0;                                 // P*H + pad
static const size_t HF_AGG  = HF_H   + (size_t)PN*HD + 2048;     // P*H + pad (also UH16)
static const size_t HF_HS   = HF_AGG + (size_t)PN*HD + 2048;     // B*H
static const size_t HF_GGC  = HF_HS  + (size_t)BB*HD;            // 3 * H*H packed
static const size_t HF_WIH  = HF_GGC + (size_t)3*HD*HD;          // H*3H packed
static const size_t HF_WHH  = HF_WIH + (size_t)HD*H3;
static const size_t HF_LWIH = HF_WHH + (size_t)HD*H3;            // H*4H packed
static const size_t HF_LWHH = HF_LWIH + (size_t)HD*H4;

__device__ __forceinline__ float sigmoidf_(float x) { return 1.0f / (1.0f + __expf(-x)); }

// ====================================================================
// WMMA GEMM: C[M,N] = A[M,128] * B[128,N]; A f16 row-major (>=16 rows of
// tail slack), B pre-packed f16 fragments, C f32. N%64==0.
// One wave: 32(M) x 64(N) strip = 32 WMMAs; B fragments shared across the
// two M-tiles. Store guarded for M%32!=0 tails (divergence only at stores).
// ====================================================================
__global__ __launch_bounds__(256) void wmma_gemm_k128(
    const _Float16* __restrict__ A, const _Float16* __restrict__ Bp,
    float* __restrict__ C, int M, int N) {
  const int lane = threadIdx.x & 31;
  const int wave = (blockIdx.x * blockDim.x + threadIdx.x) >> 5;
  const int nstrips = N >> 6;
  const int mblocks = (M + 31) >> 5;
  if (wave >= mblocks * nstrips) return;     // wave-uniform
  const int mb   = wave / nstrips;
  const int ns   = wave % nstrips;
  const int row0 = mb << 5;
  const int col0 = ns << 6;
  const int r    = lane & 15;
  const int hi   = lane >> 4;

  // ---- A fragments: 2 M-tiles x 4 K-steps (ISA 16-bit A 16x32 layout) ----
  v16h af[8];
  #pragma unroll
  for (int m = 0; m < 2; ++m) {
    const _Float16* Ar = A + (size_t)(row0 + m * 16 + r) * 128;
    #pragma unroll
    for (int f = 0; f < 4; ++f) {
      const int kb = f * 32 + hi * 8;
      v8h lo = *(const v8h*)(Ar + kb);        // K = kb..kb+7
      v8h hh = *(const v8h*)(Ar + kb + 16);   // K = kb+16..kb+23
      af[m * 4 + f] = __builtin_shufflevector(lo, hh,
          0,1,2,3,4,5,6,7,8,9,10,11,12,13,14,15);
    }
  }

  v8f acc[8] = {};
  const int ct0 = col0 >> 4;                  // first 16-wide column tile
  #pragma unroll
  for (int f = 0; f < 4; ++f) {
    v16h bfr[4];
    #pragma unroll
    for (int t = 0; t < 4; ++t)
      bfr[t] = *(const v16h*)(Bp + ((size_t)(ct0 + t) * 4 + f) * 512 + lane * 16);
    #pragma unroll
    for (int t = 0; t < 4; ++t) {
      acc[t]     = __builtin_amdgcn_wmma_f32_16x16x32_f16(
          false, af[f],     false, bfr[t], (short)0, acc[t],     false, false);
      acc[4 + t] = __builtin_amdgcn_wmma_f32_16x16x32_f16(
          false, af[4 + f], false, bfr[t], (short)0, acc[4 + t], false, false);
    }
  }

  // ---- store: C layout VGPR v -> row v + 8*hi, col = lane%16 ----
  #pragma unroll
  for (int m = 0; m < 2; ++m) {
    #pragma unroll
    for (int t = 0; t < 4; ++t) {
      const int col = col0 + t * 16 + r;
      #pragma unroll
      for (int v = 0; v < 8; ++v) {
        const int row = row0 + m * 16 + v + hi * 8;
        if (row < M) C[(size_t)row * N + col] = acc[m * 4 + t][v];
      }
    }
  }
}

static inline void launch_gemm(const _Float16* A, const _Float16* Bp, float* C,
                               int M, int N, hipStream_t s) {
  long waves = (long)((M + 31) / 32) * (N / 64);
  int grid = (int)((waves * 32 + 255) / 256);
  wmma_gemm_k128<<<grid, 256, 0, s>>>(A, Bp, C, M, N);
}

// ====================================================================
// Pack B [128,N] (f32) into WMMA-fragment-major f16.
// trans==0: B[k,n] = W[k*N + n]   (W already [K=128, N])
// trans==1: B[k,n] = W[n*128 + k] (W is [N, K=128], i.e. use W^T)
// Fragment (ct,f): 512 halves; half index = lane*16 + i,
//   k = f*32 + (i<8 ? (lane>>4)*8 + i : 16 + (lane>>4)*8 + i-8) ; n = ct*16 + (lane&15)
// ====================================================================
__global__ void pack_b_kernel(const float* __restrict__ W, _Float16* __restrict__ Bp,
                              int N, int trans) {
  int t = blockIdx.x * blockDim.x + threadIdx.x;   // over N*128 halves
  if (t >= N * 128) return;
  int i    = t & 15;
  int lane = (t >> 4) & 31;
  int f    = (t >> 9) & 3;
  int ct   = t >> 11;
  int k = f * 32 + ((lane >> 4) << 3) + (i < 8 ? i : i + 8);
  int n = ct * 16 + (lane & 15);
  float v = trans ? W[(size_t)n * 128 + k] : W[(size_t)k * N + n];
  Bp[t] = (_Float16)v;
}

// f32 -> f16 streaming convert (n % 4 == 0)
__global__ void cvt_f16_kernel(const float* __restrict__ X, _Float16* __restrict__ Y, long n4) {
  long t = (long)blockIdx.x * blockDim.x + threadIdx.x;
  if (t >= n4) return;
  float4 v = ((const float4*)X)[t];
  v4h o;
  o[0] = (_Float16)v.x; o[1] = (_Float16)v.y;
  o[2] = (_Float16)v.z; o[3] = (_Float16)v.w;
  ((v4h*)Y)[t] = o;
}

// ====================================================================
// Elementwise / reduction helper kernels
// ====================================================================
__global__ void zero_kernel(float* __restrict__ p, long n) {
  long t = (long)blockIdx.x * blockDim.x + threadIdx.x;
  if (t < n) p[t] = 0.0f;
}

__global__ void scatter_add_kernel(const float* __restrict__ msg,
                                   const int* __restrict__ src,
                                   const int* __restrict__ dst,
                                   float* __restrict__ agg, int E) {
  long t = (long)blockIdx.x * blockDim.x + threadIdx.x;
  int e  = (int)(t >> 5);                    // 32 four-channel chunks per edge
  int c4 = ((int)t & 31) << 2;
  if (e >= E) return;
  const float4 m = *(const float4*)(msg + (size_t)src[e] * HD + c4);
  float* ap = agg + (size_t)dst[e] * HD + c4;
  atomicAdd(ap + 0, m.x); atomicAdd(ap + 1, m.y);
  atomicAdd(ap + 2, m.z); atomicAdd(ap + 3, m.w);
}

__global__ void gru_gate_kernel(const float* __restrict__ gi, const float* __restrict__ gh,
                                const float* __restrict__ b_ih, const float* __restrict__ b_hh,
                                float* __restrict__ h) {
  long t = (long)blockIdx.x * blockDim.x + threadIdx.x;
  if (t >= (long)PN * HD) return;
  int i = (int)(t / HD), j = (int)(t % HD);
  const float* gir = gi + (size_t)i * H3;
  const float* ghr = gh + (size_t)i * H3;
  float ir = gir[j]        + b_ih[j];
  float iz = gir[HD + j]   + b_ih[HD + j];
  float in = gir[2*HD + j] + b_ih[2*HD + j];
  float hr = ghr[j]        + b_hh[j];
  float hz = ghr[HD + j]   + b_hh[HD + j];
  float hn = ghr[2*HD + j] + b_hh[2*HD + j];
  float r = sigmoidf_(ir + hr);
  float z = sigmoidf_(iz + hz);
  float n = tanhf(in + r * hn);
  h[t] = (1.0f - z) * n + z * h[t];
}

__global__ void gather_kernel(const float* __restrict__ src, const int* __restrict__ idx,
                              float* __restrict__ out, int nrows, int applyRelu) {
  long t = (long)blockIdx.x * blockDim.x + threadIdx.x;
  if (t >= (long)nrows * HD) return;
  int r = (int)(t / HD), j = (int)(t % HD);
  float v = src[(size_t)idx[r] * HD + j];
  out[t] = applyRelu ? fmaxf(v, 0.0f) : v;
}

__global__ void bn_stats_kernel(const float* __restrict__ X, int Mrows, int C,
                                const float* __restrict__ g, const float* __restrict__ b,
                                float* __restrict__ scale, float* __restrict__ shift) {
  int c = blockIdx.x;
  float s = 0.0f, q = 0.0f;
  for (int m = threadIdx.x; m < Mrows; m += blockDim.x) {
    float v = X[(size_t)m * C + c];
    s += v; q += v * v;
  }
  __shared__ float ss[256], sq[256];
  ss[threadIdx.x] = s; sq[threadIdx.x] = q;
  __syncthreads();
  for (int st = 128; st > 0; st >>= 1) {
    if ((int)threadIdx.x < st) { ss[threadIdx.x] += ss[threadIdx.x + st];
                                 sq[threadIdx.x] += sq[threadIdx.x + st]; }
    __syncthreads();
  }
  if (threadIdx.x == 0) {
    float mean = ss[0] / Mrows;
    float var  = sq[0] / Mrows - mean * mean;
    float sc   = g[c] * rsqrtf(var + 1e-5f);
    scale[c] = sc;
    shift[c] = b[c] - mean * sc;
  }
}

__global__ void bn_apply_kernel(float* __restrict__ X, const float* __restrict__ scale,
                                const float* __restrict__ shift, long n, int C) {
  long t = (long)blockIdx.x * blockDim.x + threadIdx.x;
  if (t >= n) return;
  int c = (int)(t % C);
  X[t] = X[t] * scale[c] + shift[c];
}

__global__ void lstm_step_kernel(const float* __restrict__ XW, const float* __restrict__ G,
                                 const float* __restrict__ b_ih, const float* __restrict__ b_hh,
                                 const int* __restrict__ length, int t,
                                 float* __restrict__ hS, float* __restrict__ cS) {
  int idx = blockIdx.x * blockDim.x + threadIdx.x;
  if (idx >= BB * HD) return;
  int b = idx >> 7, j = idx & 127;
  const float* xr = XW + ((size_t)b * LL + t) * H4;
  const float* gr = G + (size_t)b * H4;
  float ig = xr[j]        + gr[j]        + b_ih[j]        + b_hh[j];
  float fg = xr[HD + j]   + gr[HD + j]   + b_ih[HD + j]   + b_hh[HD + j];
  float gg = xr[2*HD + j] + gr[2*HD + j] + b_ih[2*HD + j] + b_hh[2*HD + j];
  float og = xr[3*HD + j] + gr[3*HD + j] + b_ih[3*HD + j] + b_hh[3*HD + j];
  float c  = sigmoidf_(fg) * cS[idx] + sigmoidf_(ig) * tanhf(gg);
  float hn = sigmoidf_(og) * tanhf(c);
  if (t < length[b]) { hS[idx] = hn; cS[idx] = c; }
}

__global__ void copy_cols_kernel(const float* __restrict__ src, float* __restrict__ dst,
                                 int Mrows, int srcC, int dstC, int dstOff) {
  long t = (long)blockIdx.x * blockDim.x + threadIdx.x;
  if (t >= (long)Mrows * srcC) return;
  int m = (int)(t / srcC), c = (int)(t % srcC);
  dst[(size_t)m * dstC + dstOff + c] = src[t];
}

__global__ void fc_kernel(const float* __restrict__ X, const float* __restrict__ W,
                          const float* __restrict__ bias, float* __restrict__ Y,
                          int Mrows, int K, int Nout, int relu_) {
  int t = blockIdx.x * blockDim.x + threadIdx.x;
  if (t >= Mrows * Nout) return;
  int m = t / Nout, o = t % Nout;
  const float* xr = X + (size_t)m * K;
  const float* wr = W + (size_t)o * K;
  float s = bias[o];
  for (int k = 0; k < K; ++k) s += xr[k] * wr[k];
  Y[t] = relu_ ? fmaxf(s, 0.0f) : s;
}

__global__ void logsoftmax_kernel(const float* __restrict__ logits, float* __restrict__ out,
                                  int Mrows, int C) {
  int m = blockIdx.x * blockDim.x + threadIdx.x;
  if (m >= Mrows) return;
  const float* r = logits + (size_t)m * C;
  float mx = r[0];
  for (int c = 1; c < C; ++c) mx = fmaxf(mx, r[c]);
  float s = 0.0f;
  for (int c = 0; c < C; ++c) s += __expf(r[c] - mx);
  float lse = mx + __logf(s);
  for (int c = 0; c < C; ++c) out[(size_t)m * C + c] = r[c] - lse;
}

// ====================================================================
// Host orchestration
// ====================================================================
static inline int gblk(long n) { return (int)((n + 255) / 256); }

extern "C" void kernel_launch(void* const* d_in, const int* in_sizes, int n_in,
                              void* d_out, int out_size, void* d_ws, size_t ws_size,
                              hipStream_t stream) {
  (void)in_sizes; (void)n_in; (void)out_size; (void)ws_size;
  const int*   user     = (const int*)  d_in[0];
  const int*   poi      = (const int*)  d_in[1];
  const int*   length   = (const int*)  d_in[2];
  const int*   topo     = (const int*)  d_in[3];   // [2,E]: src then dst
  const float* embed_w  = (const float*)d_in[4];
  const float* ggc_w    = (const float*)d_in[5];
  const float* gru_w_ih = (const float*)d_in[6];
  const float* gru_w_hh = (const float*)d_in[7];
  const float* gru_b_ih = (const float*)d_in[8];
  const float* gru_b_hh = (const float*)d_in[9];
  const float* lstm_w_ih= (const float*)d_in[10];
  const float* lstm_w_hh= (const float*)d_in[11];
  const float* lstm_b_ih= (const float*)d_in[12];
  const float* lstm_b_hh= (const float*)d_in[13];
  const float* bn_e_g   = (const float*)d_in[14];
  const float* bn_e_b   = (const float*)d_in[15];
  const float* fc1_bn1_g= (const float*)d_in[16];
  const float* fc1_bn1_b= (const float*)d_in[17];
  const float* fc1_w    = (const float*)d_in[18];
  const float* fc1_b    = (const float*)d_in[19];
  const float* fc1_bn2_g= (const float*)d_in[20];
  const float* fc1_bn2_b= (const float*)d_in[21];
  const float* fc2_bn1_g= (const float*)d_in[22];
  const float* fc2_bn1_b= (const float*)d_in[23];
  const float* fc2_w1   = (const float*)d_in[24];
  const float* fc2_b1   = (const float*)d_in[25];
  const float* fc2_bn2_g= (const float*)d_in[26];
  const float* fc2_bn2_b= (const float*)d_in[27];
  const float* fc2_w2   = (const float*)d_in[28];
  const float* fc2_b2   = (const float*)d_in[29];

  float* ws = (float*)d_ws;
  float* h    = ws + OFF_H;
  float* msg  = ws + OFF_MSG;
  float* agg  = ws + OFF_AGG;
  float* gi   = ws + OFF_GI;
  float* gh   = ws + OFF_GH;
  float* scl  = ws + OFF_SCALE;
  float* shf  = ws + OFF_SHIFT;
  float* XW   = ws + OFF_XW;
  float* G    = ws + OFF_G;
  float* UH   = ws + OFF_UH;
  float* reg  = ws + OFF_REG;
  float* poiP = ws + OFF_POI;
  float* hS   = ws + OFF_HS;
  float* cS   = ws + OFF_CS;
  float* urp  = ws + OFF_URP;
  float* urp2 = ws + OFF_URP2;
  float* upp  = ws + OFF_UPP;
  float* upp2 = ws + OFF_UPP2;
  float* logits = ws + OFF_LOG;

  _Float16* f16b  = (_Float16*)(ws + OFF_F16);
  _Float16* h16   = f16b + HF_H;
  _Float16* agg16 = f16b + HF_AGG;   // also UH16
  _Float16* hS16  = f16b + HF_HS;
  _Float16* ggcP  = f16b + HF_GGC;
  _Float16* wihP  = f16b + HF_WIH;
  _Float16* whhP  = f16b + HF_WHH;
  _Float16* lwihP = f16b + HF_LWIH;
  _Float16* lwhhP = f16b + HF_LWHH;

  const int* src = topo;
  const int* dst = topo + ED;

  // h = embed_w
  hipMemcpyAsync(h, embed_w, (size_t)PN * HD * sizeof(float),
                 hipMemcpyDeviceToDevice, stream);

  // ---- pack all B operands to WMMA fragment layout (one time) ----
  for (int s3 = 0; s3 < 3; ++s3)
    pack_b_kernel<<<gblk((long)HD * 128), 256, 0, stream>>>(
        ggc_w + (size_t)s3 * HD * HD, ggcP + (size_t)s3 * HD * HD, HD, 0);
  pack_b_kernel<<<gblk((long)H3 * 128), 256, 0, stream>>>(gru_w_ih,  wihP,  H3, 1);
  pack_b_kernel<<<gblk((long)H3 * 128), 256, 0, stream>>>(gru_w_hh,  whhP,  H3, 1);
  pack_b_kernel<<<gblk((long)H4 * 128), 256, 0, stream>>>(lstm_w_ih, lwihP, H4, 1);
  pack_b_kernel<<<gblk((long)H4 * 128), 256, 0, stream>>>(lstm_w_hh, lwhhP, H4, 1);

  // ---- GatedGraphConv: 3 propagation steps ----
  for (int step = 0; step < 3; ++step) {
    cvt_f16_kernel<<<gblk((long)PN * HD / 4), 256, 0, stream>>>(h, h16, (long)PN * HD / 4);
    launch_gemm(h16, ggcP + (size_t)step * HD * HD, msg, PN, HD, stream);
    zero_kernel<<<gblk((long)PN * HD), 256, 0, stream>>>(agg, (long)PN * HD);
    scatter_add_kernel<<<gblk((long)ED * 32), 256, 0, stream>>>(msg, src, dst, agg, ED);
    cvt_f16_kernel<<<gblk((long)PN * HD / 4), 256, 0, stream>>>(agg, agg16, (long)PN * HD / 4);
    launch_gemm(agg16, wihP, gi, PN, H3, stream);
    launch_gemm(h16,   whhP, gh, PN, H3, stream);
    gru_gate_kernel<<<gblk((long)PN * HD), 256, 0, stream>>>(gi, gh, gru_b_ih, gru_b_hh, h);
  }

  // ---- gathers (relu folded in for h) + BatchNorms ----
  gather_kernel<<<gblk((long)BL * HD), 256, 0, stream>>>(h, user, UH, BL, 1);
  gather_kernel<<<gblk((long)BB * HD), 256, 0, stream>>>(h, poi, reg, BB, 1);
  gather_kernel<<<gblk((long)BB * HD), 256, 0, stream>>>(embed_w, poi, poiP, BB, 0);

  bn_stats_kernel<<<HD, 256, 0, stream>>>(UH, BL, HD, bn_e_g, bn_e_b, scl, shf);
  bn_apply_kernel<<<gblk((long)BL * HD), 256, 0, stream>>>(UH, scl, shf, (long)BL * HD, HD);
  bn_stats_kernel<<<HD, 256, 0, stream>>>(reg, BB, HD, bn_e_g, bn_e_b, scl, shf);
  bn_apply_kernel<<<gblk((long)BB * HD), 256, 0, stream>>>(reg, scl, shf, (long)BB * HD, HD);
  bn_stats_kernel<<<HD, 256, 0, stream>>>(poiP, BB, HD, bn_e_g, bn_e_b, scl, shf);
  bn_apply_kernel<<<gblk((long)BB * HD), 256, 0, stream>>>(poiP, scl, shf, (long)BB * HD, HD);

  // ---- LSTM: XW for all timesteps via WMMA, then 50 recurrent steps ----
  cvt_f16_kernel<<<gblk((long)BL * HD / 4), 256, 0, stream>>>(UH, agg16, (long)BL * HD / 4);
  launch_gemm(agg16, lwihP, XW, BL, H4, stream);
  zero_kernel<<<gblk((long)BB * HD), 256, 0, stream>>>(hS, (long)BB * HD);
  zero_kernel<<<gblk((long)BB * HD), 256, 0, stream>>>(cS, (long)BB * HD);
  for (int t = 0; t < LL; ++t) {
    cvt_f16_kernel<<<gblk((long)BB * HD / 4), 256, 0, stream>>>(hS, hS16, (long)BB * HD / 4);
    launch_gemm(hS16, lwhhP, G, BB, H4, stream);
    lstm_step_kernel<<<gblk((long)BB * HD), 256, 0, stream>>>(
        XW, G, lstm_b_ih, lstm_b_hh, length, t, hS, cS);
  }

  // ---- fc1 ----
  copy_cols_kernel<<<gblk((long)BB * HD), 256, 0, stream>>>(hS,  urp, BB, HD, 2 * HD, 0);
  copy_cols_kernel<<<gblk((long)BB * HD), 256, 0, stream>>>(reg, urp, BB, HD, 2 * HD, HD);
  bn_stats_kernel<<<2 * HD, 256, 0, stream>>>(urp, BB, 2 * HD, fc1_bn1_g, fc1_bn1_b, scl, shf);
  bn_apply_kernel<<<gblk((long)BB * 2 * HD), 256, 0, stream>>>(urp, scl, shf, (long)BB * 2 * HD, 2 * HD);
  fc_kernel<<<gblk((long)BB * LIN), 256, 0, stream>>>(urp, fc1_w, fc1_b, urp2, BB, 2 * HD, LIN, 1);
  bn_stats_kernel<<<LIN, 256, 0, stream>>>(urp2, BB, LIN, fc1_bn2_g, fc1_bn2_b, scl, shf);
  bn_apply_kernel<<<gblk((long)BB * LIN), 256, 0, stream>>>(urp2, scl, shf, (long)BB * LIN, LIN);

  // ---- fc2 ----
  const int C2 = LIN + HD;  // 158
  copy_cols_kernel<<<gblk((long)BB * LIN), 256, 0, stream>>>(urp2, upp, BB, LIN, C2, 0);
  copy_cols_kernel<<<gblk((long)BB * HD), 256, 0, stream>>>(poiP, upp, BB, HD, C2, LIN);
  bn_stats_kernel<<<C2, 256, 0, stream>>>(upp, BB, C2, fc2_bn1_g, fc2_bn1_b, scl, shf);
  bn_apply_kernel<<<gblk((long)BB * C2), 256, 0, stream>>>(upp, scl, shf, (long)BB * C2, C2);
  fc_kernel<<<gblk((long)BB * LIN), 256, 0, stream>>>(upp, fc2_w1, fc2_b1, upp2, BB, C2, LIN, 1);
  bn_stats_kernel<<<LIN, 256, 0, stream>>>(upp2, BB, LIN, fc2_bn2_g, fc2_bn2_b, scl, shf);
  bn_apply_kernel<<<gblk((long)BB * LIN), 256, 0, stream>>>(upp2, scl, shf, (long)BB * LIN, LIN);
  fc_kernel<<<gblk((long)BB * NCLS), 256, 0, stream>>>(upp2, fc2_w2, fc2_b2, logits, BB, LIN, NCLS, 0);

  logsoftmax_kernel<<<gblk(BB), 256, 0, stream>>>(logits, (float*)d_out, BB, NCLS);
}
